// DP_Attention_12747462934739
// MI455X (gfx1250) — matrix operand order repaired and checked by reading, not compiled, not run on hardware
//
#include <hip/hip_runtime.h>

// ---------------------------------------------------------------------------
// MI455X (gfx1250) wave32 WMMA attention pipeline.
//   out  = (softmax((XQ WQ^T + bQ)(XK WK^T + bK)^T / 32) (XV WV^T + bV)) WO^T + bO
//   dist = softmax matrix, materialized fp32 once (512 MB -> dominant HBM cost)
// Matmuls: v_wmma_f32_16x16x32_bf16.  GEMM X-tiles staged to LDS with
// global_load_async_to_lds_b128 (ASYNCcnt double-buffering).
// ---------------------------------------------------------------------------

typedef __attribute__((ext_vector_type(16))) __bf16 v16bf;
typedef __attribute__((ext_vector_type(8)))  __bf16 v8bf;
typedef __attribute__((ext_vector_type(8)))  float  v8f;

union U16 { v16bf v; v8bf h[2]; };

static __device__ __forceinline__ v8f wmma_bf(v16bf a, v16bf b, v8f c) {
  return __builtin_amdgcn_wmma_f32_16x16x32_bf16(false, a, false, b, (short)0, c,
                                                 false, false);
}

static __device__ __forceinline__ v8f zero8() {
  v8f z = {0.f, 0.f, 0.f, 0.f, 0.f, 0.f, 0.f, 0.f};
  return z;
}

#define BATCH 2
#define SEQ   2048
#define DM    1024
#define NH    16
#define DH    64
#define MROWS (BATCH * SEQ)              // 4096
#define OUT_ELEMS ((size_t)MROWS * DM)   // 4194304

// ---------------------------------------------------------------------------
// GEMM: Y[4096,1024] = X * W^T + bias.  Block = 4 waves sharing one 64-row
// M-tile; each wave owns a 64-col N-tile (4x4 WMMA tiles, K steps of 32).
// X K-slice (64x32) is async-copied to LDS once per block, double-buffered.
// ---------------------------------------------------------------------------
template <bool X_IS_F32, bool OUT_F32>
__global__ __launch_bounds__(128) void gemm_xwT(const void* __restrict__ Xv,
                                                const float* __restrict__ W,
                                                const float* __restrict__ bias,
                                                void* __restrict__ Yv) {
  __shared__ __align__(16) unsigned char smem[2][10240];   // fp32: 64 rows * 160B

  const int tid  = threadIdx.x;
  const int lane = tid & 31;
  const int n16  = lane & 15;
  const int half = lane >> 4;
  const int w    = tid >> 5;
  const int mt   = blockIdx.x >> 2;                 // 0..63
  const int nt   = ((blockIdx.x & 3) << 2) | w;     // 0..15
  const int m0   = mt * 64;
  const int n0   = nt * 64;

  const unsigned sbase = (unsigned)(unsigned long long)(&smem[0][0]);

  // async stage of X[m0..m0+64) x [k0..k0+32) into smem[buf] (padded rows)
  auto issue_copy = [&](int k0, int buf) {
    if (X_IS_F32) {
      const float* X = (const float*)Xv;
#pragma unroll
      for (int j = 0; j < 4; ++j) {                 // 512 chunks of 16B
        const int c   = tid + j * 128;
        const int row = c >> 3;
        const int col = (c & 7) * 4;                // floats
        const float* g = X + (size_t)(m0 + row) * DM + k0 + col;
        const unsigned l = sbase + buf * 10240 + row * 160 + col * 4;
        asm volatile("global_load_async_to_lds_b128 %0, %1, off"
                     :: "v"(l), "v"(g) : "memory");
      }
    } else {
      const __bf16* X = (const __bf16*)Xv;
#pragma unroll
      for (int j = 0; j < 2; ++j) {                 // 256 chunks of 16B
        const int c   = tid + j * 128;
        const int row = c >> 2;
        const int col = (c & 3) * 8;                // bf16
        const __bf16* g = X + (size_t)(m0 + row) * DM + k0 + col;
        const unsigned l = sbase + buf * 10240 + row * 80 + col * 2;
        asm volatile("global_load_async_to_lds_b128 %0, %1, off"
                     :: "v"(l), "v"(g) : "memory");
      }
    }
  };

  v8f acc[4][4];
#pragma unroll
  for (int i = 0; i < 4; ++i)
#pragma unroll
    for (int j = 0; j < 4; ++j) acc[i][j] = zero8();

  issue_copy(0, 0);

  for (int i = 0; i < 32; ++i) {
    const int k0 = i * 32;
    const bool hasNext = (i + 1) < 32;
    if (hasNext) {
      issue_copy(k0 + 32, (i + 1) & 1);
      if (X_IS_F32) asm volatile("s_wait_asynccnt 4" ::: "memory");
      else          asm volatile("s_wait_asynccnt 2" ::: "memory");
    } else {
      asm volatile("s_wait_asynccnt 0" ::: "memory");
    }
    __syncthreads();                                // tile i visible to all waves

    const unsigned char* bufp = &smem[i & 1][0];
    v16bf a[4];
#pragma unroll
    for (int mi = 0; mi < 4; ++mi) {
      const int row = mi * 16 + n16;
      U16 u;
      if (X_IS_F32) {
        const float* r = (const float*)(bufp + row * 160);
        v8f f0 = *(const v8f*)(r + half * 8);
        v8f f1 = *(const v8f*)(r + 16 + half * 8);
#pragma unroll
        for (int e = 0; e < 8; ++e) {
          u.h[0][e] = (__bf16)f0[e];
          u.h[1][e] = (__bf16)f1[e];
        }
      } else {
        const __bf16* r = (const __bf16*)(bufp + row * 80);
        u.h[0] = *(const v8bf*)(r + half * 8);
        u.h[1] = *(const v8bf*)(r + 16 + half * 8);
      }
      a[mi] = u.v;
    }

    v16bf bf[4];
#pragma unroll
    for (int ni = 0; ni < 4; ++ni) {
      const float* wr = W + (size_t)(n0 + ni * 16 + n16) * DM;
      v8f f0 = *(const v8f*)(wr + k0 + half * 8);
      v8f f1 = *(const v8f*)(wr + k0 + 16 + half * 8);
      U16 u;
#pragma unroll
      for (int e = 0; e < 8; ++e) {
        u.h[0][e] = (__bf16)f0[e];
        u.h[1][e] = (__bf16)f1[e];
      }
      bf[ni] = u.v;
    }

#pragma unroll
    for (int mi = 0; mi < 4; ++mi)
#pragma unroll
      for (int ni = 0; ni < 4; ++ni) acc[mi][ni] = wmma_bf(a[mi], bf[ni], acc[mi][ni]);

    __syncthreads();                                // reads done before overwrite
  }

#pragma unroll
  for (int ni = 0; ni < 4; ++ni) {
    const int col = n0 + ni * 16 + n16;
    const float bn = bias[col];
#pragma unroll
    for (int mi = 0; mi < 4; ++mi) {
#pragma unroll
      for (int r = 0; r < 8; ++r) {
        const float vo = acc[mi][ni][r] + bn;
        const int row = m0 + mi * 16 + r + 8 * half;
        if (OUT_F32)
          ((float*)Yv)[(size_t)row * DM + col] = vo;
        else
          ((__bf16*)Yv)[(size_t)row * DM + col] = (__bf16)vo;
      }
    }
  }
}

// ---------------------------------------------------------------------------
// vT[b,h,d,l2] = v_plain[b, h*128 + l2/16, (l2%16)*64 + d]  (raw-reshape map)
// ---------------------------------------------------------------------------
__global__ __launch_bounds__(256) void vtrans(const __bf16* __restrict__ vp,
                                              __bf16* __restrict__ vT) {
  const int idx = blockIdx.x * blockDim.x + threadIdx.x;
  if (idx >= BATCH * NH * DH * SEQ) return;
  const int bh = idx >> 17;
  const int r  = idx & 131071;
  const int d  = r >> 11;
  const int l2 = r & 2047;
  const int b = bh >> 4, h = bh & 15;
  const size_t src = (size_t)(b * SEQ + h * 128 + (l2 >> 4)) * DM + (l2 & 15) * DH + d;
  vT[idx] = vp[src];
}

// ---------------------------------------------------------------------------
// Pass 1: per 32-row tile of (b,h): streaming row max / sumexp of
// scores = q k^T / 32.  2 M-tiles per wave share every k B-fragment.
// ---------------------------------------------------------------------------
__global__ __launch_bounds__(128) void attn_stats(const __bf16* __restrict__ qb,
                                                  const __bf16* __restrict__ kb,
                                                  float* __restrict__ ml) {
  const int lane = threadIdx.x & 31;
  const int n16  = lane & 15;
  const int half = lane >> 4;
  const int gw   = blockIdx.x * 4 + (threadIdx.x >> 5);
  const int bh   = gw >> 6;            // 64 tiles of 32 rows per head
  const int i0   = (gw & 63) << 5;
  const int b    = bh >> 4, h = bh & 15;

  v16bf aq[2][2];
#pragma unroll
  for (int mi = 0; mi < 2; ++mi) {
    const __bf16* qrow =
        qb + (size_t)(b * SEQ + h * 128 + (i0 >> 4) + mi) * DM + n16 * DH;
#pragma unroll
    for (int t = 0; t < 2; ++t) {
      U16 u;
      u.h[0] = *(const v8bf*)(qrow + t * 32 + half * 16);
      u.h[1] = *(const v8bf*)(qrow + t * 32 + half * 16 + 8);
      aq[mi][t] = u.v;
    }
  }

  float mr[2][8], lr[2][8];
#pragma unroll
  for (int mi = 0; mi < 2; ++mi)
#pragma unroll
    for (int r = 0; r < 8; ++r) { mr[mi][r] = -1e30f; lr[mi][r] = 0.f; }

  const size_t kbase = (size_t)(b * SEQ + h * 128) * DM;
  for (int j0 = 0; j0 < SEQ; j0 += 64) {
    v8f s[2][4];
#pragma unroll
    for (int nt = 0; nt < 4; ++nt) {
      const __bf16* krow = kb + kbase + (size_t)((j0 >> 4) + nt) * DM + n16 * DH;
      v16bf b0 = *(const v16bf*)(krow + half * 16);
      v16bf b1 = *(const v16bf*)(krow + 32 + half * 16);
#pragma unroll
      for (int mi = 0; mi < 2; ++mi) {
        v8f c = zero8();
        c = wmma_bf(aq[mi][0], b0, c);
        c = wmma_bf(aq[mi][1], b1, c);
        s[mi][nt] = c * 0.03125f;     // / (dh/2)
      }
    }
#pragma unroll
    for (int mi = 0; mi < 2; ++mi) {
#pragma unroll
      for (int r = 0; r < 8; ++r) {
        float mx = s[mi][0][r];
#pragma unroll
        for (int nt = 1; nt < 4; ++nt) mx = fmaxf(mx, s[mi][nt][r]);
#pragma unroll
        for (int msk = 1; msk < 16; msk <<= 1) mx = fmaxf(mx, __shfl_xor(mx, msk, 32));
        const float nm = fmaxf(mr[mi][r], mx);
        const float corr = __expf(mr[mi][r] - nm);
        float ps = 0.f;
#pragma unroll
        for (int nt = 0; nt < 4; ++nt) ps += __expf(s[mi][nt][r] - nm);
#pragma unroll
        for (int msk = 1; msk < 16; msk <<= 1) ps += __shfl_xor(ps, msk, 32);
        lr[mi][r] = lr[mi][r] * corr + ps;
        mr[mi][r] = nm;
      }
    }
  }

  if (n16 == 0) {
#pragma unroll
    for (int mi = 0; mi < 2; ++mi)
#pragma unroll
      for (int r = 0; r < 8; ++r) {
        const int row = i0 + mi * 16 + r + 8 * half;
        float* p = ml + (size_t)(bh * SEQ + row) * 2;
        p[0] = mr[mi][r];
        p[1] = lr[mi][r];
      }
  }
}

// ---------------------------------------------------------------------------
// Pass 2: recompute scores, p = exp(s-m)/l, write dist fp32 once,
// LDS round-trip p (C-layout -> A-layout bf16), fuse attn += p*v (WMMA),
// scatter attn into plain bf16 layout for the output GEMM.
// ---------------------------------------------------------------------------
__global__ __launch_bounds__(128) void attn_apply(const __bf16* __restrict__ qb,
                                                  const __bf16* __restrict__ kb,
                                                  const __bf16* __restrict__ vT,
                                                  const float* __restrict__ ml,
                                                  float* __restrict__ dist,
                                                  __bf16* __restrict__ attnb) {
  __shared__ __align__(32) __bf16 ptile[4][32][64];   // per-wave 32x64 p tile

  const int lane = threadIdx.x & 31;
  const int n16  = lane & 15;
  const int half = lane >> 4;
  const int w    = threadIdx.x >> 5;
  const int gw   = blockIdx.x * 4 + w;
  const int bh   = gw >> 6;
  const int i0   = (gw & 63) << 5;
  const int b    = bh >> 4, h = bh & 15;

  v16bf aq[2][2];
#pragma unroll
  for (int mi = 0; mi < 2; ++mi) {
    const __bf16* qrow =
        qb + (size_t)(b * SEQ + h * 128 + (i0 >> 4) + mi) * DM + n16 * DH;
#pragma unroll
    for (int t = 0; t < 2; ++t) {
      U16 u;
      u.h[0] = *(const v8bf*)(qrow + t * 32 + half * 16);
      u.h[1] = *(const v8bf*)(qrow + t * 32 + half * 16 + 8);
      aq[mi][t] = u.v;
    }
  }

  float mr[2][8], li[2][8];
#pragma unroll
  for (int mi = 0; mi < 2; ++mi)
#pragma unroll
    for (int r = 0; r < 8; ++r) {
      const int row = i0 + mi * 16 + r + 8 * half;
      const float* p = ml + (size_t)(bh * SEQ + row) * 2;
      mr[mi][r] = p[0];
      li[mi][r] = 1.0f / p[1];
    }

  v8f oacc[2][4];
#pragma unroll
  for (int mi = 0; mi < 2; ++mi)
#pragma unroll
    for (int dt = 0; dt < 4; ++dt) oacc[mi][dt] = zero8();

  const size_t kbase = (size_t)(b * SEQ + h * 128) * DM;
  float* drow = dist + (size_t)bh * SEQ * SEQ;

  for (int j0 = 0; j0 < SEQ; j0 += 64) {
    v8f s[2][4];
#pragma unroll
    for (int nt = 0; nt < 4; ++nt) {
      const __bf16* krow = kb + kbase + (size_t)((j0 >> 4) + nt) * DM + n16 * DH;
      v16bf b0 = *(const v16bf*)(krow + half * 16);
      v16bf b1 = *(const v16bf*)(krow + 32 + half * 16);
#pragma unroll
      for (int mi = 0; mi < 2; ++mi) {
        v8f c = zero8();
        c = wmma_bf(aq[mi][0], b0, c);
        c = wmma_bf(aq[mi][1], b1, c);
        s[mi][nt] = c * 0.03125f;
      }
    }
#pragma unroll
    for (int mi = 0; mi < 2; ++mi) {
#pragma unroll
      for (int nt = 0; nt < 4; ++nt) {
#pragma unroll
        for (int r = 0; r < 8; ++r) {
          const float pv = __expf(s[mi][nt][r] - mr[mi][r]) * li[mi][r];
          const int row = i0 + mi * 16 + r + 8 * half;
          drow[(size_t)row * SEQ + j0 + nt * 16 + n16] = pv;           // dist fp32
          ptile[w][mi * 16 + r + 8 * half][nt * 16 + n16] = (__bf16)pv;
        }
      }
    }
    // p A-fragments from LDS (row-major -> 16x32 A layout)
    v16bf pa[2][2];
#pragma unroll
    for (int mi = 0; mi < 2; ++mi)
#pragma unroll
      for (int t = 0; t < 2; ++t) {
        U16 u;
        u.h[0] = *(const v8bf*)&ptile[w][mi * 16 + n16][t * 32 + half * 16];
        u.h[1] = *(const v8bf*)&ptile[w][mi * 16 + n16][t * 32 + half * 16 + 8];
        pa[mi][t] = u.v;
      }
    // vT B-fragments shared across both M-tiles
#pragma unroll
    for (int dt = 0; dt < 4; ++dt) {
      const __bf16* vrow = vT + ((size_t)bh * DH + dt * 16 + n16) * SEQ + j0;
      v16bf b0 = *(const v16bf*)(vrow + half * 16);
      v16bf b1 = *(const v16bf*)(vrow + 32 + half * 16);
#pragma unroll
      for (int mi = 0; mi < 2; ++mi) {
        oacc[mi][dt] = wmma_bf(pa[mi][0], b0, oacc[mi][dt]);
        oacc[mi][dt] = wmma_bf(pa[mi][1], b1, oacc[mi][dt]);
      }
    }
  }

#pragma unroll
  for (int mi = 0; mi < 2; ++mi) {
    const size_t pr = (size_t)(b * SEQ + h * 128 + (i0 >> 4) + mi) * DM;
#pragma unroll
    for (int dt = 0; dt < 4; ++dt)
#pragma unroll
      for (int r = 0; r < 8; ++r)
        attnb[pr + (size_t)(r + 8 * half) * DH + dt * 16 + n16] =
            (__bf16)oacc[mi][dt][r];
  }
}

// ---------------------------------------------------------------------------
// launcher
// ---------------------------------------------------------------------------
extern "C" void kernel_launch(void* const* d_in, const int* in_sizes, int n_in,
                              void* d_out, int out_size, void* d_ws, size_t ws_size,
                              hipStream_t stream) {
  const float* Q  = (const float*)d_in[0];
  const float* K  = (const float*)d_in[1];
  const float* V  = (const float*)d_in[2];
  const float* WQ = (const float*)d_in[3];
  const float* bQ = (const float*)d_in[4];
  const float* WK = (const float*)d_in[5];
  const float* bK = (const float*)d_in[6];
  const float* WV = (const float*)d_in[7];
  const float* bV = (const float*)d_in[8];
  const float* WO = (const float*)d_in[9];
  const float* bO = (const float*)d_in[10];

  char* ws = (char*)d_ws;
  const size_t PLANE = (size_t)MROWS * DM * sizeof(__bf16);  // 8 MB
  __bf16* qb  = (__bf16*)(ws + 0 * PLANE);
  __bf16* kb  = (__bf16*)(ws + 1 * PLANE);
  __bf16* vb  = (__bf16*)(ws + 2 * PLANE);
  __bf16* vT  = (__bf16*)(ws + 3 * PLANE);
  __bf16* atb = (__bf16*)(ws + 4 * PLANE);
  float*  mlb = (float*)(ws + 5 * PLANE);   // B*H*SEQ*2 floats = 512 KB

  float* out  = (float*)d_out;
  float* dist = out + OUT_ELEMS;

  const dim3 gGemm(256), bGemm(128);        // 1024 waves
  gemm_xwT<true, false><<<gGemm, bGemm, 0, stream>>>((const void*)Q, WQ, bQ, (void*)qb);
  gemm_xwT<true, false><<<gGemm, bGemm, 0, stream>>>((const void*)K, WK, bK, (void*)kb);
  gemm_xwT<true, false><<<gGemm, bGemm, 0, stream>>>((const void*)V, WV, bV, (void*)vb);

  vtrans<<<dim3((BATCH * NH * DH * SEQ) / 256), dim3(256), 0, stream>>>(vb, vT);

  const dim3 gAttn(512), bAttn(128);        // 2048 tiles of 32 rows, 4 waves/block
  attn_stats<<<gAttn, bAttn, 0, stream>>>(qb, kb, mlb);
  attn_apply<<<gAttn, bAttn, 0, stream>>>(qb, kb, vT, mlb, dist, atb);

  gemm_xwT<false, true><<<gGemm, bGemm, 0, stream>>>((const void*)atb, WO, bO, (void*)out);
}
